// VidHRFormerBlockDecNAR_55911884259737
// MI455X (gfx1250) — compile-verified, matrix operand order checked by reading, not run on hardware
//
#include <hip/hip_runtime.h>
#include <hip/hip_bf16.h>
#include <math.h>

typedef __bf16 bf16_t;
typedef __attribute__((ext_vector_type(16))) __bf16 v16bf;
typedef __attribute__((ext_vector_type(8)))  float  v8f;

static constexpr int    NB   = 8, TT = 10, HH = 16, WW = 16, CC = 512;
static constexpr int    NHD  = 8, HID = 2048;
static constexpr size_t MTOK = (size_t)NB * TT * HH * WW;   // 20480 tokens
static constexpr size_t MC   = MTOK * CC;                   // 10,485,760

#define DEV static __device__ __forceinline__

DEV float geluf(float x) { return 0.5f * x * (1.0f + erff(x * 0.70710678118654752f)); }

// Build a 16x32 bf16 WMMA fragment from an LDS row (k-contiguous, 16B aligned).
// ISA layout: lanes 0-15 -> K{0..7,16..23}, lanes 16-31 -> K{8..15,24..31}.
DEV v16bf load_frag(const bf16_t* row, int hi) {
    union { uint4 q[2]; v16bf v; } u;
    u.q[0] = *(const uint4*)(row + hi * 8);
    u.q[1] = *(const uint4*)(row + 16 + hi * 8);
    return u.v;
}

// CDNA5 async copy: 16 bytes global -> LDS, no VGPR staging, tracked by ASYNCcnt.
// LDS byte address = low 32 bits of the flat shared-aperture address (ISA 10.2).
DEV void async_cp16(bf16_t* lds_dst, const bf16_t* gsrc) {
    unsigned int loff = (unsigned int)(unsigned long long)(uintptr_t)lds_dst;
    unsigned long long ga = (unsigned long long)(uintptr_t)gsrc;
    asm volatile("global_load_async_to_lds_b128 %0, %1, off"
                 :: "v"(loff), "v"(ga) : "memory");
}

// ---------------------------------------------------------------------------
// Generic GEMM:  out[M][N] = A[M][K](bf16) @ B[N][K]^T(bf16) + bias
// Block tile 128x64x32, 128 threads (4 waves); each wave owns 32 rows x 64 cols.
// Double-buffered LDS fed by global_load_async_to_lds_b128 (ASYNCcnt) so the
// tile-(k+1) prefetch overlaps the tile-k WMMAs.
// ---------------------------------------------------------------------------
#define GP 40   // padded LDS row stride (bf16 elems): 80 B, conflict-free b128 reads

__global__ __launch_bounds__(128) void gemm_bf16_wmma(
    const bf16_t* __restrict__ A, const bf16_t* __restrict__ B,
    const float* __restrict__ bias, const float* res,
    float* outF, bf16_t* outB, int M, int N, int K)
{
    __shared__ __align__(16) bf16_t As[2][128 * GP];
    __shared__ __align__(16) bf16_t Bs[2][64 * GP];

    const int n0 = blockIdx.x * 64, m0 = blockIdx.y * 128;
    const int tid = threadIdx.x;
    const int wave = tid >> 5, lane = tid & 31, lm = lane & 15, hi = lane >> 4;

    // per-thread staging chunks: A = 512 chunks (128 rows x 4), B = 256 chunks
    int ar[4], ac[4], br[2], bc[2];
    #pragma unroll
    for (int i = 0; i < 4; ++i) { int id = tid + 128 * i; ar[i] = id >> 2; ac[i] = id & 3; }
    #pragma unroll
    for (int i = 0; i < 2; ++i) { int id = tid + 128 * i; br[i] = id >> 2; bc[i] = id & 3; }

    v8f acc[2][4] = {};
    const int nk = K >> 5;

    // prefetch tile 0 into buffer 0
    #pragma unroll
    for (int i = 0; i < 4; ++i)
        async_cp16(&As[0][ar[i] * GP + ac[i] * 8], A + (size_t)(m0 + ar[i]) * K + ac[i] * 8);
    #pragma unroll
    for (int i = 0; i < 2; ++i)
        async_cp16(&Bs[0][br[i] * GP + bc[i] * 8], B + (size_t)(n0 + br[i]) * K + bc[i] * 8);

    for (int kk = 0; kk < nk; ++kk) {
        const int buf = kk & 1;
        if (kk + 1 < nk) {
            const int k1 = (kk + 1) << 5;
            #pragma unroll
            for (int i = 0; i < 4; ++i)
                async_cp16(&As[buf ^ 1][ar[i] * GP + ac[i] * 8],
                           A + (size_t)(m0 + ar[i]) * K + k1 + ac[i] * 8);
            #pragma unroll
            for (int i = 0; i < 2; ++i)
                async_cp16(&Bs[buf ^ 1][br[i] * GP + bc[i] * 8],
                           B + (size_t)(n0 + br[i]) * K + k1 + bc[i] * 8);
            asm volatile("s_wait_asynccnt 0x6" ::: "memory");   // tile kk done; 6 in flight
        } else {
            asm volatile("s_wait_asynccnt 0x0" ::: "memory");
        }
        __syncthreads();                                        // all waves' tile kk visible

        v16bf a0 = load_frag(&As[buf][(wave * 32 + lm) * GP], hi);
        v16bf a1 = load_frag(&As[buf][(wave * 32 + 16 + lm) * GP], hi);
        #pragma unroll
        for (int j = 0; j < 4; ++j) {
            v16bf b = load_frag(&Bs[buf][(j * 16 + lm) * GP], hi);
            acc[0][j] = __builtin_amdgcn_wmma_f32_16x16x32_bf16(false, a0, false, b,
                                                                (short)0, acc[0][j], false, false);
            acc[1][j] = __builtin_amdgcn_wmma_f32_16x16x32_bf16(false, a1, false, b,
                                                                (short)0, acc[1][j], false, false);
        }
        __syncthreads();   // buf fully consumed before iter kk+1 prefetches into it
    }

    #pragma unroll
    for (int s = 0; s < 2; ++s) {
        const int mrow = m0 + wave * 32 + s * 16 + hi * 8;
        #pragma unroll
        for (int j = 0; j < 4; ++j) {
            int col = n0 + j * 16 + lm;
            float bv = bias ? bias[col] : 0.0f;
            #pragma unroll
            for (int v = 0; v < 8; ++v) {
                size_t idx = (size_t)(mrow + v) * N + col;
                float val = acc[s][j][v] + bv;
                if (outB) outB[idx] = (bf16_t)val;
                else      outF[idx] = val + (res ? res[idx] : 0.0f);
            }
        }
    }
}

// ---------------------------------------------------------------------------
// Flash attention: Q,K,V,O in bf16 [BH][L|S][64].  Block = 4 waves, 64 q-rows.
// Online softmax in f32; scores & P*V via bf16 WMMA; V staged transposed.
// ---------------------------------------------------------------------------
#define FP 72   // padded LDS row stride (144 B)

__global__ __launch_bounds__(128) void flash_attn_wmma(
    const bf16_t* __restrict__ Q, const bf16_t* __restrict__ K,
    const bf16_t* __restrict__ V, bf16_t* __restrict__ O,
    int L, int S, float scale)
{
    __shared__ __align__(16) bf16_t Qs[64 * FP];
    __shared__ __align__(16) bf16_t Ks[64 * FP];
    __shared__ __align__(16) bf16_t Vt[64 * FP];   // transposed: [dim][key]
    __shared__ __align__(16) bf16_t Ps[64 * FP];

    const int bh = blockIdx.y;
    const int q0 = blockIdx.x * 64;
    const bf16_t* Qb = Q + (size_t)bh * L * 64;
    const bf16_t* Kb = K + (size_t)bh * S * 64;
    const bf16_t* Vb = V + (size_t)bh * S * 64;
    bf16_t*       Ob = O + (size_t)bh * L * 64;

    const int tid = threadIdx.x;
    const int wave = tid >> 5, lane = tid & 31, lm = lane & 15, hi = lane >> 4;
    const uint4 zz = make_uint4(0u, 0u, 0u, 0u);

    // stage Q tile (zero-padded past L)
    #pragma unroll
    for (int i = 0; i < 4; ++i) {
        int id = tid * 4 + i;
        int r = id >> 3, c = id & 7;
        uint4 val = zz;
        if (q0 + r < L) val = *(const uint4*)(Qb + (size_t)(q0 + r) * 64 + c * 8);
        *(uint4*)(Qs + r * FP + c * 8) = val;
    }
    __syncthreads();
    v16bf aq0 = load_frag(Qs + (wave * 16 + lm) * FP, hi);
    v16bf aq1 = load_frag(Qs + (wave * 16 + lm) * FP + 32, hi);

    v8f acc[4] = {};
    float rmax[8], rsum[8];
    #pragma unroll
    for (int v = 0; v < 8; ++v) { rmax[v] = -3.0e38f; rsum[v] = 0.0f; }

    for (int s0 = 0; s0 < S; s0 += 64) {
        __syncthreads();
        #pragma unroll
        for (int i = 0; i < 4; ++i) {                // stage K chunk + V^T chunk
            int id = tid * 4 + i;
            int r = id >> 3, c = id & 7;
            uint4 kv = zz, vv = zz;
            if (s0 + r < S) {
                kv = *(const uint4*)(Kb + (size_t)(s0 + r) * 64 + c * 8);
                vv = *(const uint4*)(Vb + (size_t)(s0 + r) * 64 + c * 8);
            }
            *(uint4*)(Ks + r * FP + c * 8) = kv;
            union { uint4 q; bf16_t e[8]; } u; u.q = vv;
            #pragma unroll
            for (int d2 = 0; d2 < 8; ++d2) Vt[(c * 8 + d2) * FP + r] = u.e[d2];
        }
        __syncthreads();

        float p[4][8];
        #pragma unroll
        for (int j = 0; j < 4; ++j) {                // S_ij = Q(16x64) . K_j^T
            v8f s = {};
            v16bf bk0 = load_frag(Ks + (j * 16 + lm) * FP, hi);
            v16bf bk1 = load_frag(Ks + (j * 16 + lm) * FP + 32, hi);
            s = __builtin_amdgcn_wmma_f32_16x16x32_bf16(false, aq0, false, bk0, (short)0, s, false, false);
            s = __builtin_amdgcn_wmma_f32_16x16x32_bf16(false, aq1, false, bk1, (short)0, s, false, false);
            int col = s0 + j * 16 + lm;
            #pragma unroll
            for (int v = 0; v < 8; ++v) p[j][v] = (col < S) ? s[v] * scale : -3.0e38f;
        }

        #pragma unroll
        for (int v = 0; v < 8; ++v) {                // online softmax per row
            float mx = fmaxf(fmaxf(p[0][v], p[1][v]), fmaxf(p[2][v], p[3][v]));
            #pragma unroll
            for (int x = 1; x < 16; x <<= 1) mx = fmaxf(mx, __shfl_xor(mx, x, 32));
            float mnew = fmaxf(rmax[v], mx);
            float corr = __expf(rmax[v] - mnew);
            float lsum = 0.0f;
            #pragma unroll
            for (int j = 0; j < 4; ++j) { float e = __expf(p[j][v] - mnew); p[j][v] = e; lsum += e; }
            #pragma unroll
            for (int x = 1; x < 16; x <<= 1) lsum += __shfl_xor(lsum, x, 32);
            rsum[v] = rsum[v] * corr + lsum;
            rmax[v] = mnew;
            #pragma unroll
            for (int j = 0; j < 4; ++j) acc[j][v] *= corr;
        }

        #pragma unroll
        for (int j = 0; j < 4; ++j)                  // re-stage P in A-frag layout
            #pragma unroll
            for (int v = 0; v < 8; ++v)
                Ps[(wave * 16 + hi * 8 + v) * FP + j * 16 + lm] = (bf16_t)p[j][v];
        __syncthreads();

        v16bf ap0 = load_frag(Ps + (wave * 16 + lm) * FP, hi);
        v16bf ap1 = load_frag(Ps + (wave * 16 + lm) * FP + 32, hi);
        #pragma unroll
        for (int j = 0; j < 4; ++j) {                // O += P(16x64) . V(64x16)
            v16bf bv0 = load_frag(Vt + (j * 16 + lm) * FP, hi);
            v16bf bv1 = load_frag(Vt + (j * 16 + lm) * FP + 32, hi);
            acc[j] = __builtin_amdgcn_wmma_f32_16x16x32_bf16(false, ap0, false, bv0, (short)0, acc[j], false, false);
            acc[j] = __builtin_amdgcn_wmma_f32_16x16x32_bf16(false, ap1, false, bv1, (short)0, acc[j], false, false);
        }
    }

    #pragma unroll
    for (int v = 0; v < 8; ++v) {
        float inv = 1.0f / rsum[v];
        int row = q0 + wave * 16 + hi * 8 + v;
        if (row < L) {
            #pragma unroll
            for (int j = 0; j < 4; ++j)
                Ob[(size_t)row * 64 + j * 16 + lm] = (bf16_t)(acc[j][v] * inv);
        }
    }
}

// ---------------------------------------------------------------------------
// Elementwise / layout kernels
// ---------------------------------------------------------------------------
__global__ __launch_bounds__(256) void layernorm_k(
    const float* __restrict__ x, const float* __restrict__ g,
    const float* __restrict__ b, float* __restrict__ y, int ntok)
{
    int wv = blockIdx.x * (blockDim.x >> 5) + (threadIdx.x >> 5);
    if (wv >= ntok) return;
    int lane = threadIdx.x & 31;
    const float* xr = x + (size_t)wv * CC;
    float4 e[4]; float s = 0.0f;
    #pragma unroll
    for (int i = 0; i < 4; ++i) {
        e[i] = *(const float4*)(xr + lane * 4 + i * 128);
        s += e[i].x + e[i].y + e[i].z + e[i].w;
    }
    #pragma unroll
    for (int x2 = 1; x2 < 32; x2 <<= 1) s += __shfl_xor(s, x2, 32);
    float mean = s * (1.0f / CC), vs = 0.0f;
    #pragma unroll
    for (int i = 0; i < 4; ++i) {
        float dx = e[i].x - mean, dy = e[i].y - mean, dz = e[i].z - mean, dw = e[i].w - mean;
        vs += dx * dx + dy * dy + dz * dz + dw * dw;
    }
    #pragma unroll
    for (int x2 = 1; x2 < 32; x2 <<= 1) vs += __shfl_xor(vs, x2, 32);
    float rst = rsqrtf(vs * (1.0f / CC) + 1e-5f);
    float* yr = y + (size_t)wv * CC;
    #pragma unroll
    for (int i = 0; i < 4; ++i) {
        int c = lane * 4 + i * 128;
        float4 o;
        o.x = (e[i].x - mean) * rst * g[c + 0] + b[c + 0];
        o.y = (e[i].y - mean) * rst * g[c + 1] + b[c + 1];
        o.z = (e[i].z - mean) * rst * g[c + 2] + b[c + 2];
        o.w = (e[i].w - mean) * rst * g[c + 3] + b[c + 3];
        *(float4*)(yr + c) = o;
    }
}

DEV void tok_decode(int tok, int& n, int& t, int& h, int& w) {
    w = tok & 15; h = (tok >> 4) & 15;
    int rest = tok >> 8; t = rest % TT; n = rest / TT;
}

__global__ void add_pos_cvt_k(const float* __restrict__ src, const float* __restrict__ pos,
                              bf16_t* __restrict__ out, int mode)
{
    size_t i = (size_t)blockIdx.x * blockDim.x + threadIdx.x;
    if (i >= MC) return;
    int tok = (int)(i >> 9), c = (int)(i & 511);
    float v = src[i];
    if (pos) {
        int n, t, h, w; tok_decode(tok, n, t, h, w);
        int cls = (mode == 0) ? ((h & 7) * 8 + (w & 7))
                : (mode == 1) ? t
                              : ((t * 8 + (h & 7)) * 8 + (w & 7));
        v += pos[(size_t)cls * CC + c];
    }
    out[i] = (bf16_t)v;
}

DEV void head_map(int tok, int mode, int& l, int& b) {
    int n, t, h, w; tok_decode(tok, n, t, h, w);
    if (mode == 0)      { l = (h & 7) * 8 + (w & 7); b = ((n * TT + t) * 2 + (h >> 3)) * 2 + (w >> 3); }
    else if (mode == 1) { l = t;                     b = (n * 16 + h) * 16 + w; }
    else                { l = (t * 8 + (h & 7)) * 8 + (w & 7); b = (n * 2 + (h >> 3)) * 2 + (w >> 3); }
}

__global__ void scatter_heads_k(const bf16_t* __restrict__ src, int stride, int off,
                                bf16_t* __restrict__ dst, int mode, int Ld)
{
    size_t i = (size_t)blockIdx.x * blockDim.x + threadIdx.x;
    if (i >= MC) return;
    int tok = (int)(i >> 9), c = (int)(i & 511);
    int head = c >> 6, d = c & 63, l, b;
    head_map(tok, mode, l, b);
    dst[(((size_t)(b * NHD + head)) * Ld + l) * 64 + d] = src[(size_t)tok * stride + off + c];
}

__global__ void merge_heads_k(const bf16_t* __restrict__ src, bf16_t* __restrict__ dst,
                              int mode, int Ld)
{
    size_t i = (size_t)blockIdx.x * blockDim.x + threadIdx.x;
    if (i >= MC) return;
    int tok = (int)(i >> 9), c = (int)(i & 511);
    int head = c >> 6, d = c & 63, l, b;
    head_map(tok, mode, l, b);
    dst[i] = src[(((size_t)(b * NHD + head)) * Ld + l) * 64 + d];
}

__global__ void bn_gelu_k(const float* __restrict__ in, float* out,
                          const float* __restrict__ g, const float* __restrict__ b,
                          const float* __restrict__ m, const float* __restrict__ v,
                          int Ch, size_t n, int add_res)
{
    size_t i = (size_t)blockIdx.x * blockDim.x + threadIdx.x;
    if (i >= n) return;
    int c = (int)(i % Ch);
    float x = (in[i] - m[c]) * rsqrtf(v[c] + 1e-5f) * g[c] + b[c];
    x = geluf(x);
    if (add_res) out[i] += x; else out[i] = x;
}

__global__ void dwconv_bn_gelu_k(const float* __restrict__ in, const float* __restrict__ wgt,
                                 const float* __restrict__ bias,
                                 const float* __restrict__ g, const float* __restrict__ b,
                                 const float* __restrict__ m, const float* __restrict__ v,
                                 bf16_t* __restrict__ out)
{
    size_t i = (size_t)blockIdx.x * blockDim.x + threadIdx.x;
    if (i >= MTOK * HID) return;
    int c = (int)(i % HID);
    size_t p = i / HID;
    int wv = (int)(p & 15), hv = (int)((p >> 4) & 15);
    size_t img = p >> 8;
    float acc = bias[c];
    #pragma unroll
    for (int ky = -1; ky <= 1; ++ky)
        #pragma unroll
        for (int kx = -1; kx <= 1; ++kx) {
            int hh = hv + ky, ww2 = wv + kx;
            if (hh >= 0 && hh < 16 && ww2 >= 0 && ww2 < 16)
                acc += in[(((img * 16 + hh) * 16 + ww2)) * HID + c] * wgt[c * 9 + (ky + 1) * 3 + (kx + 1)];
        }
    acc = (acc - m[c]) * rsqrtf(v[c] + 1e-5f) * g[c] + b[c];
    out[i] = (bf16_t)geluf(acc);
}

__global__ void gelu_cvt_k(const float* __restrict__ in, bf16_t* __restrict__ out, size_t n)
{
    size_t i = (size_t)blockIdx.x * blockDim.x + threadIdx.x;
    if (i < n) out[i] = (bf16_t)geluf(in[i]);
}

__global__ void cvt_bf16_k(const float* __restrict__ in, bf16_t* __restrict__ out, size_t n)
{
    size_t i = (size_t)blockIdx.x * blockDim.x + threadIdx.x;
    if (i < n) out[i] = (bf16_t)in[i];
}

// ---------------------------------------------------------------------------
// Host orchestration
// ---------------------------------------------------------------------------
static inline int g1(size_t n) { return (int)((n + 255) / 256); }

static inline void gemm(hipStream_t s, const bf16_t* A, const bf16_t* B, const float* bias,
                        const float* res, float* outF, bf16_t* outB, int M, int N, int K)
{
    dim3 grid(N / 64, M / 128);
    gemm_bf16_wmma<<<grid, 128, 0, s>>>(A, B, bias, res, outF, outB, M, N, K);
}

extern "C" void kernel_launch(void* const* d_in, const int* in_sizes, int n_in,
                              void* d_out, int out_size, void* d_ws, size_t ws_size,
                              hipStream_t stream)
{
    (void)in_sizes; (void)n_in; (void)out_size; (void)ws_size;
    const float* q_in  = (const float*)d_in[0];
    const float* mem   = (const float*)d_in[1];
    const float* pos_l = (const float*)d_in[2];
    const float* pos_t = (const float*)d_in[3];
    const float* pos_ts = (const float*)d_in[4];
    const float *ln_g[5], *ln_b[5];
    for (int i = 0; i < 5; ++i) { ln_g[i] = (const float*)d_in[5 + 2 * i]; ln_b[i] = (const float*)d_in[6 + 2 * i]; }
    const float* sa_w_in = (const float*)d_in[15]; const float* sa_b_in = (const float*)d_in[16];
    const float* sa_w_o  = (const float*)d_in[17]; const float* sa_b_o  = (const float*)d_in[18];
    const float* ta_w_in = (const float*)d_in[19]; const float* ta_b_in = (const float*)d_in[20];
    const float* ta_w_o  = (const float*)d_in[21]; const float* ta_b_o  = (const float*)d_in[22];
    const float* ca_w_in = (const float*)d_in[23]; const float* ca_b_in = (const float*)d_in[24];
    const float* ca_w_o  = (const float*)d_in[25]; const float* ca_b_o  = (const float*)d_in[26];
    const float* fc1_w = (const float*)d_in[27]; const float* fc1_b = (const float*)d_in[28];
    const float* dw_w  = (const float*)d_in[29]; const float* dw_b  = (const float*)d_in[30];
    const float* fc2_w = (const float*)d_in[31]; const float* fc2_b = (const float*)d_in[32];
    const float* bn1[4] = {(const float*)d_in[33], (const float*)d_in[34], (const float*)d_in[35], (const float*)d_in[36]};
    const float* bn2[4] = {(const float*)d_in[37], (const float*)d_in[38], (const float*)d_in[39], (const float*)d_in[40]};
    const float* bn3[4] = {(const float*)d_in[41], (const float*)d_in[42], (const float*)d_in[43], (const float*)d_in[44]};
    const float* lin1_w = (const float*)d_in[45]; const float* lin1_b = (const float*)d_in[46];
    const float* lin2_w = (const float*)d_in[47]; const float* lin2_b = (const float*)d_in[48];

    float* xo = (float*)d_out;

    // workspace arena
    char* ws = (char*)d_ws;
    float*  LN = (float*)ws;                         // Mtok x C  f32
    float*  F0 = (float*)(ws + MC * 4);              // Mtok x HID f32
    bf16_t* H0 = (bf16_t*)(ws + MC * 4 + MC * 16);   // 4 slots of Mtok x C bf16
    bf16_t* H1 = H0 + 4 * MC;
    bf16_t* H2 = H1 + 4 * MC;
    bf16_t* wb = H2 + 4 * MC;
    bf16_t* wb_sa_in = wb;                 wb += (size_t)3 * CC * CC;
    bf16_t* wb_sa_o  = wb;                 wb += (size_t)CC * CC;
    bf16_t* wb_ta_in = wb;                 wb += (size_t)3 * CC * CC;
    bf16_t* wb_ta_o  = wb;                 wb += (size_t)CC * CC;
    bf16_t* wb_ca_in = wb;                 wb += (size_t)3 * CC * CC;
    bf16_t* wb_ca_o  = wb;                 wb += (size_t)CC * CC;
    bf16_t* wb_fc1   = wb;                 wb += (size_t)HID * CC;
    bf16_t* wb_fc2   = wb;                 wb += (size_t)CC * HID;
    bf16_t* wb_l1    = wb;                 wb += (size_t)HID * CC;
    bf16_t* wb_l2    = wb;

    const float SCALE = 0.125f;   // 1/sqrt(64)
    const int M = (int)MTOK;

    // weights -> bf16
    auto cvtw = [&](const float* s, bf16_t* d, size_t n) {
        cvt_bf16_k<<<g1(n), 256, 0, stream>>>(s, d, n);
    };
    cvtw(sa_w_in, wb_sa_in, (size_t)3 * CC * CC); cvtw(sa_w_o, wb_sa_o, (size_t)CC * CC);
    cvtw(ta_w_in, wb_ta_in, (size_t)3 * CC * CC); cvtw(ta_w_o, wb_ta_o, (size_t)CC * CC);
    cvtw(ca_w_in, wb_ca_in, (size_t)3 * CC * CC); cvtw(ca_w_o, wb_ca_o, (size_t)CC * CC);
    cvtw(fc1_w, wb_fc1, (size_t)HID * CC); cvtw(fc2_w, wb_fc2, (size_t)CC * HID);
    cvtw(lin1_w, wb_l1, (size_t)HID * CC); cvtw(lin2_w, wb_l2, (size_t)CC * HID);

    hipMemcpyAsync(xo, q_in, MC * 4, hipMemcpyDeviceToDevice, stream);

    // ---------------- 1) windowed spatial self-attention ----------------
    layernorm_k<<<(M + 7) / 8, 256, 0, stream>>>(xo, ln_g[0], ln_b[0], LN, M);
    add_pos_cvt_k<<<g1(MC), 256, 0, stream>>>(LN, pos_l, H0, 0);               // q/k input
    add_pos_cvt_k<<<g1(MC), 256, 0, stream>>>(LN, nullptr, H0 + MC, 0);        // v input
    gemm(stream, H0,      wb_sa_in,             sa_b_in,        nullptr, nullptr, H1,          M, 1024, CC);
    gemm(stream, H0 + MC, wb_sa_in + 1024 * CC, sa_b_in + 1024, nullptr, nullptr, H1 + 2 * MC, M, 512,  CC);
    scatter_heads_k<<<g1(MC), 256, 0, stream>>>(H1,          1024, 0,   H2,          0, 64);
    scatter_heads_k<<<g1(MC), 256, 0, stream>>>(H1,          1024, 512, H2 + MC,     0, 64);
    scatter_heads_k<<<g1(MC), 256, 0, stream>>>(H1 + 2 * MC, 512,  0,   H2 + 2 * MC, 0, 64);
    flash_attn_wmma<<<dim3(1, 2560), 128, 0, stream>>>(H2, H2 + MC, H2 + 2 * MC, H0 + 2 * MC, 64, 64, SCALE);
    merge_heads_k<<<g1(MC), 256, 0, stream>>>(H0 + 2 * MC, H0 + 3 * MC, 0, 64);
    gemm(stream, H0 + 3 * MC, wb_sa_o, sa_b_o, xo, xo, nullptr, M, 512, CC);

    // ---------------- 2) spatial FFN (MlpDWBN) ----------------
    layernorm_k<<<(M + 7) / 8, 256, 0, stream>>>(xo, ln_g[1], ln_b[1], LN, M);
    add_pos_cvt_k<<<g1(MC), 256, 0, stream>>>(LN, nullptr, H0, 0);
    gemm(stream, H0, wb_fc1, fc1_b, nullptr, F0, nullptr, M, HID, CC);
    bn_gelu_k<<<g1(4 * MC), 256, 0, stream>>>(F0, F0, bn1[0], bn1[1], bn1[2], bn1[3], HID, 4 * MC, 0);
    dwconv_bn_gelu_k<<<g1(4 * MC), 256, 0, stream>>>(F0, dw_w, dw_b, bn2[0], bn2[1], bn2[2], bn2[3], H0);
    gemm(stream, H0, wb_fc2, fc2_b, nullptr, LN, nullptr, M, 512, HID);
    bn_gelu_k<<<g1(MC), 256, 0, stream>>>(LN, xo, bn3[0], bn3[1], bn3[2], bn3[3], 512, MC, 1);

    // ---------------- 3) temporal MHSA ----------------
    layernorm_k<<<(M + 7) / 8, 256, 0, stream>>>(xo, ln_g[2], ln_b[2], LN, M);
    add_pos_cvt_k<<<g1(MC), 256, 0, stream>>>(LN, pos_t, H0, 1);
    add_pos_cvt_k<<<g1(MC), 256, 0, stream>>>(LN, nullptr, H0 + MC, 1);
    gemm(stream, H0,      wb_ta_in,             ta_b_in,        nullptr, nullptr, H1,          M, 1024, CC);
    gemm(stream, H0 + MC, wb_ta_in + 1024 * CC, ta_b_in + 1024, nullptr, nullptr, H1 + 2 * MC, M, 512,  CC);
    scatter_heads_k<<<g1(MC), 256, 0, stream>>>(H1,          1024, 0,   H2,          1, 10);
    scatter_heads_k<<<g1(MC), 256, 0, stream>>>(H1,          1024, 512, H2 + MC,     1, 10);
    scatter_heads_k<<<g1(MC), 256, 0, stream>>>(H1 + 2 * MC, 512,  0,   H2 + 2 * MC, 1, 10);
    flash_attn_wmma<<<dim3(1, 16384), 128, 0, stream>>>(H2, H2 + MC, H2 + 2 * MC, H0 + 2 * MC, 10, 10, SCALE);
    merge_heads_k<<<g1(MC), 256, 0, stream>>>(H0 + 2 * MC, H0 + 3 * MC, 1, 10);
    gemm(stream, H0 + 3 * MC, wb_ta_o, ta_b_o, xo, xo, nullptr, M, 512, CC);

    // ---------------- 4) temporal-spatial cross-attention ----------------
    layernorm_k<<<(M + 7) / 8, 256, 0, stream>>>(xo, ln_g[3], ln_b[3], LN, M);
    add_pos_cvt_k<<<g1(MC), 256, 0, stream>>>(LN,  pos_ts + (size_t)TT * 64 * CC, H0,          2); // q
    add_pos_cvt_k<<<g1(MC), 256, 0, stream>>>(mem, pos_ts,                        H0 + MC,     2); // k
    add_pos_cvt_k<<<g1(MC), 256, 0, stream>>>(mem, nullptr,                       H0 + 2 * MC, 2); // v
    gemm(stream, H0,          wb_ca_in,                       ca_b_in,        nullptr, nullptr, H1,          M, 512, CC);
    gemm(stream, H0 + MC,     wb_ca_in + (size_t)CC * CC,     ca_b_in + 512,  nullptr, nullptr, H1 + MC,     M, 512, CC);
    gemm(stream, H0 + 2 * MC, wb_ca_in + (size_t)2 * CC * CC, ca_b_in + 1024, nullptr, nullptr, H1 + 2 * MC, M, 512, CC);
    scatter_heads_k<<<g1(MC), 256, 0, stream>>>(H1,          512, 0, H2,          2, 640);
    scatter_heads_k<<<g1(MC), 256, 0, stream>>>(H1 + MC,     512, 0, H2 + MC,     2, 640);
    scatter_heads_k<<<g1(MC), 256, 0, stream>>>(H1 + 2 * MC, 512, 0, H2 + 2 * MC, 2, 640);
    flash_attn_wmma<<<dim3(10, 256), 128, 0, stream>>>(H2, H2 + MC, H2 + 2 * MC, H0 + 2 * MC, 640, 640, SCALE);
    merge_heads_k<<<g1(MC), 256, 0, stream>>>(H0 + 2 * MC, H0 + 3 * MC, 2, 640);
    gemm(stream, H0 + 3 * MC, wb_ca_o, ca_b_o, xo, xo, nullptr, M, 512, CC);

    // ---------------- 5) final FFN ----------------
    layernorm_k<<<(M + 7) / 8, 256, 0, stream>>>(xo, ln_g[4], ln_b[4], LN, M);
    add_pos_cvt_k<<<g1(MC), 256, 0, stream>>>(LN, nullptr, H0, 0);
    gemm(stream, H0, wb_l1, lin1_b, nullptr, F0, nullptr, M, HID, CC);
    gelu_cvt_k<<<g1(4 * MC), 256, 0, stream>>>(F0, H0, 4 * MC);
    gemm(stream, H0, wb_l2, lin2_b, xo, xo, nullptr, M, 512, HID);
}
